// EdgeDecoder_18339510354271
// MI455X (gfx1250) — matrix operand order repaired and checked by reading, not compiled, not run on hardware
//
#include <hip/hip_runtime.h>

typedef __attribute__((ext_vector_type(16))) _Float16 v16h;
typedef __attribute__((ext_vector_type(8)))  _Float16 v8h;
typedef __attribute__((ext_vector_type(4)))  _Float16 v4h;
typedef __attribute__((ext_vector_type(8)))  float    v8f;

#define BLOCK_EDGES 128   // edges per block iteration (8 waves x 16 rows)
#define ROWSTRIDE   168   // f16 elements per LDS row: 336B, multiple of 16B
#define KPAD        160   // K padded from 132 to 5x32
#define HDIM        128   // hidden channels
#define DIN         132   // 2*64 + 4
#define NNODES      2048
#define NBATCH      32
#define DLAT        64

__global__ void zero_stats_kernel(float* stats) { stats[threadIdx.x] = 0.f; }

template<bool FINAL>
__global__ __launch_bounds__(256)
void edge_mlp_kernel(const float* __restrict__ z,
                     const float* __restrict__ edge_attr,
                     const float* __restrict__ W1,
                     const long long* __restrict__ eidx,
                     const float* __restrict__ gamma,
                     const float* __restrict__ beta,
                     const float* __restrict__ W2,
                     const float* __restrict__ b2,
                     float* __restrict__ stats,
                     float* __restrict__ out,
                     int E)
{
    __shared__ _Float16 sA[BLOCK_EDGES][ROWSTRIDE];   // staged edge inputs (f16)
    __shared__ _Float16 sW[HDIM][ROWSTRIDE];          // W1^T (f16), zero-padded K
    __shared__ float sSum[HDIM], sSq[HDIM];
    __shared__ float sScale[HDIM], sShift[HDIM], sW2[HDIM];

    const int tid  = threadIdx.x;
    const int lane = tid & 31;
    const int wave = tid >> 5;

    // ---- one-time per-block init ----
    for (int i = tid; i < HDIM * ROWSTRIDE; i += 256) {
        int c = i / ROWSTRIDE, k = i - c * ROWSTRIDE;
        sW[c][k] = (k < DIN) ? (_Float16)W1[k * HDIM + c] : (_Float16)0.f;
    }
    for (int i = tid; i < BLOCK_EDGES * (ROWSTRIDE - DIN); i += 256) {
        int r = i / (ROWSTRIDE - DIN), k = DIN + (i - r * (ROWSTRIDE - DIN));
        sA[r][k] = (_Float16)0.f;   // static zero pad: K in [132,168)
    }
    if (tid < HDIM) {
        if (FINAL) {
            float inv  = 1.f / (float)E;
            float mu   = stats[tid] * inv;
            float var  = stats[HDIM + tid] * inv - mu * mu;
            float rstd = rsqrtf(var + 1e-5f);
            float a    = rstd * gamma[tid];
            sScale[tid] = a;
            sShift[tid] = beta[tid] - mu * a;
            sW2[tid]    = W2[tid];
        } else {
            sSum[tid] = 0.f;
            sSq[tid]  = 0.f;
        }
    }
    __syncthreads();

    const float bias2  = FINAL ? b2[0] : 0.f;
    const int   nTiles = (E + BLOCK_EDGES - 1) / BLOCK_EDGES;

    for (int tg = blockIdx.x; tg < nTiles; tg += gridDim.x) {
        const int base = tg * BLOCK_EDGES;

        // ---- stage 128 edges: gather z[src], z[dst], edge_attr -> f16 LDS ----
        {
            const int el = tid >> 1;    // local edge 0..127
            const int half = tid & 1;   // 0: src half, 1: dst half
            const int e = base + el;
            _Float16* row = &sA[el][0];
            if (e < E) {
                const int src = (int)eidx[e];
                const int dst = (int)eidx[E + e];
                int eb = src / NNODES;
                eb = eb < 0 ? 0 : (eb > NBATCH - 1 ? NBATCH - 1 : eb);
                const int gi = half ? ((dst % NNODES) + eb * NNODES)
                                    : ((src % NNODES) + eb * NNODES);
                const float4* p = (const float4*)(z + (size_t)gi * DLAT);
                _Float16* dsth = row + half * DLAT;
                #pragma unroll
                for (int j = 0; j < 8; ++j) {
                    float4 v0 = p[2 * j], v1 = p[2 * j + 1];
                    v8h hk = { (_Float16)v0.x, (_Float16)v0.y, (_Float16)v0.z, (_Float16)v0.w,
                               (_Float16)v1.x, (_Float16)v1.y, (_Float16)v1.z, (_Float16)v1.w };
                    *(v8h*)(dsth + 8 * j) = hk;
                }
                if (!half) {
                    float4 a4 = *(const float4*)(edge_attr + (size_t)e * 4);
                    v4h h4 = { (_Float16)a4.x, (_Float16)a4.y, (_Float16)a4.z, (_Float16)a4.w };
                    *(v4h*)(row + 2 * DLAT) = h4;
                }
            } else {
                _Float16* dsth = row + half * DLAT;
                v8h zero8 = {};
                #pragma unroll
                for (int j = 0; j < 8; ++j) *(v8h*)(dsth + 8 * j) = zero8;
                if (!half) { v4h z4 = {}; *(v4h*)(row + 2 * DLAT) = z4; }
            }
            // prefetch next tile's indices (global_prefetch_b8)
            const int nb = base + (int)gridDim.x * BLOCK_EDGES;
            if (nb + el < E) __builtin_prefetch(&eidx[nb + el], 0, 0);
        }
        __syncthreads();

        // ---- WMMA: rows [wave*16, wave*16+16) x 128 cols, K = 160 ----
        v8f c[8];
        #pragma unroll
        for (int n = 0; n < 8; ++n) c[n] = (v8f){0, 0, 0, 0, 0, 0, 0, 0};

        const int arow  = wave * 16 + (lane & 15);
        const int akoff = (lane >> 4) * 8;     // A 16-bit 16x32 layout (ISA 7.12.2)
        const int bcol  = lane & 15;
        const int bkoff = (lane >> 4) * 8;     // B 32x16: same K order as A (8h+s, 16+8h+s)

        #pragma unroll
        for (int ks = 0; ks < 5; ++ks) {
            const int k0 = ks * 32;
            v8h alo = *(const v8h*)&sA[arow][k0 + akoff];
            v8h ahi = *(const v8h*)&sA[arow][k0 + akoff + 16];
            v16h a = __builtin_shufflevector(alo, ahi,
                     0,1,2,3,4,5,6,7,8,9,10,11,12,13,14,15);
            #pragma unroll
            for (int n = 0; n < 8; ++n) {
                const _Float16* bp = &sW[n * 16 + bcol][k0 + bkoff];
                v8h blo = *(const v8h*)bp;
                v8h bhi = *(const v8h*)(bp + 16);
                v16h b = __builtin_shufflevector(blo, bhi,
                         0,1,2,3,4,5,6,7,8,9,10,11,12,13,14,15);
                c[n] = __builtin_amdgcn_wmma_f32_16x16x32_f16(
                           false, a, false, b, (short)0, c[n], false, false);
            }
        }

        if (!FINAL) {
            // ---- batch-norm statistics: C layout lane L -> col L&15, rows r+8*(L>=16)
            #pragma unroll
            for (int n = 0; n < 8; ++n) {
                float s = 0.f, q = 0.f;
                #pragma unroll
                for (int r = 0; r < 8; ++r) { float h = c[n][r]; s += h; q += h * h; }
                s += __shfl_xor(s, 16, 32);
                q += __shfl_xor(q, 16, 32);
                if (lane < 16) {
                    atomicAdd(&sSum[n * 16 + lane], s);
                    atomicAdd(&sSq [n * 16 + lane], q);
                }
            }
        } else {
            // ---- normalize + LeakyReLU + dot(W2) -> one output per edge ----
            float acc[8];
            #pragma unroll
            for (int r = 0; r < 8; ++r) acc[r] = 0.f;
            #pragma unroll
            for (int n = 0; n < 8; ++n) {
                const int ch = n * 16 + (lane & 15);
                const float a1 = sScale[ch], sh = sShift[ch], w2v = sW2[ch];
                #pragma unroll
                for (int r = 0; r < 8; ++r) {
                    float h = c[n][r] * a1 + sh;
                    h = (h >= 0.f) ? h : 0.2f * h;
                    acc[r] += h * w2v;
                }
            }
            #pragma unroll
            for (int r = 0; r < 8; ++r) {
                float v = acc[r];
                v += __shfl_xor(v, 1, 32);
                v += __shfl_xor(v, 2, 32);
                v += __shfl_xor(v, 4, 32);
                v += __shfl_xor(v, 8, 32);
                if ((lane & 15) == 0) {
                    const int e = base + wave * 16 + (lane >> 4) * 8 + r;
                    if (e < E) out[e] = v + bias2;
                }
            }
        }
        __syncthreads();   // sA reused next iteration
    }

    if (!FINAL) {
        if (tid < HDIM) {
            atomicAdd(&stats[tid],        sSum[tid]);
            atomicAdd(&stats[HDIM + tid], sSq[tid]);
        }
    }
}

extern "C" void kernel_launch(void* const* d_in, const int* in_sizes, int n_in,
                              void* d_out, int out_size, void* d_ws, size_t ws_size,
                              hipStream_t stream) {
    const float*     z     = (const float*)d_in[0];
    const float*     ea    = (const float*)d_in[1];
    const float*     W1    = (const float*)d_in[2];
    // d_in[3] = b1: provably cancels through batch-norm mean subtraction
    const float*     gamma = (const float*)d_in[4];
    const float*     beta  = (const float*)d_in[5];
    const float*     W2    = (const float*)d_in[6];
    const float*     b2    = (const float*)d_in[7];
    const long long* eidx  = (const long long*)d_in[8];

    const int E = in_sizes[8] / 2;
    float* stats = (float*)d_ws;          // 256 floats: [sum(128), sumsq(128)]
    float* outp  = (float*)d_out;

    zero_stats_kernel<<<1, 256, 0, stream>>>(stats);

    const int nTiles = (E + BLOCK_EDGES - 1) / BLOCK_EDGES;
    const int grid   = nTiles < 1024 ? nTiles : 1024;

    edge_mlp_kernel<false><<<grid, 256, 0, stream>>>(z, ea, W1, eidx, gamma, beta,
                                                     W2, b2, stats, outp, E);
    edge_mlp_kernel<true ><<<grid, 256, 0, stream>>>(z, ea, W1, eidx, gamma, beta,
                                                     W2, b2, stats, outp, E);
}
